// FiLMLearnableKernel_46566035423349
// MI455X (gfx1250) — matrix operand order repaired and checked by reading, not compiled, not run on hardware
//
#include <hip/hip_runtime.h>
#include <math.h>

typedef __attribute__((ext_vector_type(16))) _Float16 v16h;
typedef __attribute__((ext_vector_type(8)))  float    v8f;

#define B_ 4
#define N_ 256
#define M_ 512
#define E_ 256
#define H_ 64

// ---- branch-free transcendentals on HW trans ops (v_exp_f32 / v_rcp_f32) ----

__device__ __forceinline__ float fast_rcp(float x) {
    return __builtin_amdgcn_rcpf(x);
}

// Abramowitz & Stegun 7.1.26 minimax erf, |err| ~ 1.5e-7, branch-free.
__device__ __forceinline__ float erf_fast(float x) {
    float ax = fabsf(x);
    float t  = fast_rcp(fmaf(0.3275911f, ax, 1.0f));
    float p  = fmaf(t, 1.061405429f, -1.453152027f);
    p = fmaf(t, p, 1.421413741f);
    p = fmaf(t, p, -0.284496736f);
    p = fmaf(t, p, 0.254829592f);
    p = p * t;
    float e = __expf(-ax * ax);           // single v_exp_f32
    float r = fmaf(-p, e, 1.0f);
    return copysignf(r, x);
}

__device__ __forceinline__ float gelu_exact(float x) {
    // exact (erf) GELU, matching jax.nn.gelu(approximate=False)
    return 0.5f * x * (1.0f + erf_fast(x * 0.70710678118654752440f));
}

__device__ __forceinline__ float sigmoid_fast(float x) {
    return fast_rcp(1.0f + __expf(-x));
}

// ---------------------------------------------------------------- kernel 0
__global__ void init_max_kernel(unsigned* maxbits) { *maxbits = 0u; }

// ---------------------------------------------------------------- kernel 1
// max over all pairwise distances (P=2). distances >= 0 so float bit pattern
// compares like uint -> atomicMax on uint bits.
__global__ void dist_max_kernel(const float* __restrict__ z,
                                const float* __restrict__ mu,
                                unsigned* __restrict__ maxbits) {
    __shared__ float red[256];
    int idx = blockIdx.x * blockDim.x + threadIdx.x;   // [0, B*N*M)
    int m  = idx % M_;
    int bn = idx / M_;
    float dx = z[m * 2 + 0] - mu[bn * 2 + 0];
    float dy = z[m * 2 + 1] - mu[bn * 2 + 1];
    float d  = sqrtf(dx * dx + dy * dy);
    red[threadIdx.x] = d;
    __syncthreads();
    for (int s = 128; s > 0; s >>= 1) {
        if (threadIdx.x < s)
            red[threadIdx.x] = fmaxf(red[threadIdx.x], red[threadIdx.x + s]);
        __syncthreads();
    }
    if (threadIdx.x == 0) atomicMax(maxbits, __float_as_uint(red[0]));
}

// ---------------------------------------------------------------- kernel 2
// FiLM generator: film[bn][0:128] = Linear(H,2H)(gelu(Linear(E,H)(emb)))
__global__ void film_kernel(const float* __restrict__ emb,
                            const float* __restrict__ w1, const float* __restrict__ b1,
                            const float* __restrict__ w2, const float* __restrict__ b2,
                            float* __restrict__ film) {
    __shared__ float e[E_];
    __shared__ float h[H_];
    int row = blockIdx.x;            // bn in [0, B*N)
    int t   = threadIdx.x;           // 0..63
    for (int i = t; i < E_; i += 64) e[i] = emb[row * E_ + i];
    __syncthreads();
    float acc = b1[t];
    #pragma unroll 4
    for (int k = 0; k < E_; ++k) acc += e[k] * w1[t * E_ + k];
    h[t] = gelu_exact(acc);
    __syncthreads();
    for (int g = t; g < 2 * H_; g += 64) {
        float a = b2[g];
        #pragma unroll 4
        for (int k = 0; k < H_; ++k) a += h[k] * w2[g * H_ + k];
        film[row * 2 * H_ + g] = a;
    }
}

// ---------------------------------------------------------------- kernel 3
// Kernel net with FiLM modulation using f16 WMMA with f32 accumulation.
// One block per (b,n); 8 waves; each wave processes 16-row tiles of M.
__global__ void __launch_bounds__(256)
kernelnet_wmma(const float* __restrict__ z,
               const float* __restrict__ mu,
               const float* __restrict__ kn_w0, const float* __restrict__ kn_b0,
               const float* __restrict__ kn_w2, const float* __restrict__ kn_b2,
               const float* __restrict__ kn_w4, const float* __restrict__ kn_b4,
               const float* __restrict__ film,
               const unsigned* __restrict__ maxbits,
               float* __restrict__ out) {
    __shared__ __align__(32) _Float16 w2h[H_ * H_];   // kn_w2 as f16, [g][h]
    __shared__ float scaleS[H_], shiftS[H_], w0S[H_], b0S[H_], b2S[H_], w4S[H_];

    int bn  = blockIdx.x;
    int tid = threadIdx.x;

    for (int i = tid; i < H_ * H_; i += 256) w2h[i] = (_Float16)kn_w2[i];
    if (tid < H_) {
        scaleS[tid] = film[bn * 2 * H_ + tid];
        shiftS[tid] = film[bn * 2 * H_ + H_ + tid];
        w0S[tid]    = kn_w0[tid];
        b0S[tid]    = kn_b0[tid];
        b2S[tid]    = kn_b2[tid];
        w4S[tid]    = kn_w4[tid];
    }
    __syncthreads();

    float mu0 = mu[bn * 2 + 0], mu1 = mu[bn * 2 + 1];
    float inv = fast_rcp(__uint_as_float(*maxbits) + 1e-8f);
    float b4  = kn_b4[0];

    int lane   = tid & 31;
    int wave   = tid >> 5;
    int laneM  = lane & 15;
    int hiHalf = lane >> 4;          // 0 for lanes 0-15, 1 for lanes 16-31

    // B fragments (32x16 f16), invariant across tiles/waves: [colTile][kGroup].
    // B element (K=h, N=g) = kn_w2[g][h]; lanes 0-15 hold K = kg*32 + 0..15,
    // lanes 16-31 hold K = kg*32 + 16..31, column = lane & 15.
    v16h bfrag[4][2];
    #pragma unroll
    for (int ct = 0; ct < 4; ++ct) {
        int g = ct * 16 + laneM;
        #pragma unroll
        for (int kg = 0; kg < 2; ++kg) {
            int hbase = kg * 32 + hiHalf * 16;          // 32-byte aligned
            bfrag[ct][kg] = *(const v16h*)&w2h[g * H_ + hbase];
        }
    }

    for (int t = wave; t < M_ / 16; t += 8) {
        int   m  = t * 16 + laneM;
        float dx = z[m * 2 + 0] - mu0;
        float dy = z[m * 2 + 1] - mu1;
        float nd = sqrtf(dx * dx + dy * dy) * inv;

        // Layer 1 + FiLM + GELU, packed directly into the A-fragment layout:
        // lanes 0-15 row=laneM, halves j: K = j;
        // lanes 16-31: +16. Second fragment covers K=32..63.
        v16h a0, a1;
        #pragma unroll
        for (int j = 0; j < 16; ++j) {
            int k = j + hiHalf * 16;
            float x0 = fmaf(nd, w0S[k], b0S[k]);
            a0[j] = (_Float16)gelu_exact(fmaf(scaleS[k], x0, shiftS[k]));
            int k2 = k + 32;
            float x1 = fmaf(nd, w0S[k2], b0S[k2]);
            a1[j] = (_Float16)gelu_exact(fmaf(scaleS[k2], x1, shiftS[k2]));
        }

        v8f c0 = {}, c1 = {}, c2 = {}, c3 = {};
        c0 = __builtin_amdgcn_wmma_f32_16x16x32_f16(false, a0, false, bfrag[0][0], (short)0, c0, false, false);
        c0 = __builtin_amdgcn_wmma_f32_16x16x32_f16(false, a1, false, bfrag[0][1], (short)0, c0, false, false);
        c1 = __builtin_amdgcn_wmma_f32_16x16x32_f16(false, a0, false, bfrag[1][0], (short)0, c1, false, false);
        c1 = __builtin_amdgcn_wmma_f32_16x16x32_f16(false, a1, false, bfrag[1][1], (short)0, c1, false, false);
        c2 = __builtin_amdgcn_wmma_f32_16x16x32_f16(false, a0, false, bfrag[2][0], (short)0, c2, false, false);
        c2 = __builtin_amdgcn_wmma_f32_16x16x32_f16(false, a1, false, bfrag[2][1], (short)0, c2, false, false);
        c3 = __builtin_amdgcn_wmma_f32_16x16x32_f16(false, a0, false, bfrag[3][0], (short)0, c3, false, false);
        c3 = __builtin_amdgcn_wmma_f32_16x16x32_f16(false, a1, false, bfrag[3][1], (short)0, c3, false, false);

        // Epilogue: C/D layout -> N = laneM + 16*ct, M-row = v + 8*hiHalf.
        // Second FiLM+GELU, then dot with kn_w4 across g (cross-lane + cross-ct).
        float acc[8];
        #pragma unroll
        for (int v = 0; v < 8; ++v) acc[v] = 0.0f;
        #pragma unroll
        for (int v = 0; v < 8; ++v) {
            { int g = 0 * 16 + laneM; float y = gelu_exact(fmaf(scaleS[g], c0[v] + b2S[g], shiftS[g])); acc[v] = fmaf(y, w4S[g], acc[v]); }
            { int g = 1 * 16 + laneM; float y = gelu_exact(fmaf(scaleS[g], c1[v] + b2S[g], shiftS[g])); acc[v] = fmaf(y, w4S[g], acc[v]); }
            { int g = 2 * 16 + laneM; float y = gelu_exact(fmaf(scaleS[g], c2[v] + b2S[g], shiftS[g])); acc[v] = fmaf(y, w4S[g], acc[v]); }
            { int g = 3 * 16 + laneM; float y = gelu_exact(fmaf(scaleS[g], c3[v] + b2S[g], shiftS[g])); acc[v] = fmaf(y, w4S[g], acc[v]); }
        }
        // Reduce across the 16 lanes of each half (xor masks < 16 stay in-half).
        #pragma unroll
        for (int v = 0; v < 8; ++v) {
            float s = acc[v];
            s += __shfl_xor(s, 1, 32);
            s += __shfl_xor(s, 2, 32);
            s += __shfl_xor(s, 4, 32);
            s += __shfl_xor(s, 8, 32);
            acc[v] = s;
        }
        if (laneM == 0) {   // lane 0 writes rows 0-7, lane 16 writes rows 8-15
            int rowBase = t * 16 + hiHalf * 8;
            #pragma unroll
            for (int v = 0; v < 8; ++v) {
                out[bn * M_ + rowBase + v] = sigmoid_fast(acc[v] + b4);
            }
        }
    }
}

// ---------------------------------------------------------------- launch
extern "C" void kernel_launch(void* const* d_in, const int* in_sizes, int n_in,
                              void* d_out, int out_size, void* d_ws, size_t ws_size,
                              hipStream_t stream) {
    const float* z      = (const float*)d_in[0];
    const float* mu     = (const float*)d_in[1];
    const float* emb    = (const float*)d_in[2];
    const float* fg_w1  = (const float*)d_in[3];
    const float* fg_b1  = (const float*)d_in[4];
    const float* fg_w2  = (const float*)d_in[5];
    const float* fg_b2  = (const float*)d_in[6];
    const float* kn_w0  = (const float*)d_in[7];
    const float* kn_b0  = (const float*)d_in[8];
    const float* kn_w2  = (const float*)d_in[9];
    const float* kn_b2  = (const float*)d_in[10];
    const float* kn_w4  = (const float*)d_in[11];
    const float* kn_b4  = (const float*)d_in[12];
    float* out = (float*)d_out;

    unsigned* maxbits = (unsigned*)d_ws;
    float*    film    = (float*)((char*)d_ws + 256);   // [B*N][2H] f32 = 512 KB

    init_max_kernel<<<1, 1, 0, stream>>>(maxbits);
    dist_max_kernel<<<(B_ * N_ * M_) / 256, 256, 0, stream>>>(z, mu, maxbits);
    film_kernel<<<B_ * N_, 64, 0, stream>>>(emb, fg_w1, fg_b1, fg_w2, fg_b2, film);
    kernelnet_wmma<<<B_ * N_, 256, 0, stream>>>(z, mu, kn_w0, kn_b0, kn_w2, kn_b2,
                                                kn_w4, kn_b4, film, maxbits, out);
}
